// HPCrnn_67542655696951
// MI455X (gfx1250) — matrix operand order repaired and checked by reading, not compiled, not run on hardware
//
#include <hip/hip_runtime.h>
#include <math.h>

typedef _Float16 h16;
typedef __attribute__((ext_vector_type(16))) _Float16 v16h;
typedef __attribute__((ext_vector_type(8)))  _Float16 v8h;
typedef __attribute__((ext_vector_type(8)))  float    v8f;

// Global-address-space views so weight loads are statically global_load (not flat)
typedef __attribute__((address_space(1))) const h16  gh16;
typedef __attribute__((address_space(1))) const v16h gv16h;

#define BS      32768
#define STEPS   100
#define ROWS    128      // batch rows per block (8 waves x 16 rows)
#define NT      16       // N tiles of 16 (256 cols)
#define KT      8        // K tiles of 32 (256 deep)

// Optimization barrier: redefines the pointer each call so LICM cannot hoist
// the loop-invariant weight loads out of the 100-step loop (which would spill
// ~2048 VGPRs of fragments to scratch). Barrier is applied to a u64 in an
// SGPR pair, then cast to an addrspace(1) pointer so loads stay global_load
// with saddr + 24-bit immediate addressing.
__device__ __forceinline__ gh16* opaque(const h16* p) {
  unsigned long long u = (unsigned long long)p;
  asm volatile("" : "+s"(u));
  return (gh16*)u;
}

__device__ __forceinline__ float clip01(float x) {
  return fminf(fmaxf(x, 0.0f), 1.0f);
}

// Swizzled LDS index: rows of 256 halfs, 16-byte chunks XOR-permuted by row
// so A-fragment b128 reads (16 lanes, 16 consecutive rows, same chunk) hit
// 16 distinct bank quads instead of one.
__device__ __forceinline__ int sidx(int row, int col) {
  int c = col >> 3;
  return row * 256 + (((c ^ (row & 15)) << 3) | (col & 7));
}

// Load the 8 K-fragments (16x32 f16 A tiles) for this wave's 16 rows from LDS.
// A layout: lanes 0-15 carry M=lane with K-chunks {4k+0, 4k+1}; lanes 16-31
// carry M=lane-16 with K-chunks {4k+2, 4k+3}.
__device__ __forceinline__ void load_a_frags(const h16* S, int wrow, int m, int hi,
                                             v16h a[KT]) {
  const int rsw = (wrow + m) & 15;
  const h16* arow = S + (wrow + m) * 256;
#pragma unroll
  for (int k = 0; k < KT; ++k) {
    v8h lo = *(const v8h*)(arow + ((((k << 2) + 2 * hi)     ^ rsw) << 3));
    v8h hh = *(const v8h*)(arow + ((((k << 2) + 2 * hi + 1) ^ rsw) << 3));
    v16h av;
#pragma unroll
    for (int i = 0; i < 8; ++i) { av[i] = lo[i]; av[i + 8] = hh[i]; }
    a[k] = av;
  }
}

// One 16x16 output tile of A[16x256] * W[256x256]. All 8 B fragments are
// loaded up front (one load clause, drained by the WMMA chain with
// progressively smaller s_wait_loadcnt) instead of load->wait(0)->wmma
// serialization. 'off' = lane base (m*256+hi*16) + nt*4096; the (nt,k)
// variation folds into the 24-bit instruction offset.
__device__ __forceinline__ v8f gemm_tile(const v16h a[KT], gh16* WT, int off) {
  v16h b[KT];
#pragma unroll
  for (int k = 0; k < KT; ++k)
    b[k] = *(gv16h*)(WT + off + k * 32);
  v8f acc;
#pragma unroll
  for (int r = 0; r < 8; ++r) acc[r] = 0.0f;
#pragma unroll
  for (int k = 0; k < KT; ++k)
    acc = __builtin_amdgcn_wmma_f32_16x16x32_f16(false, a[k], false, b[k],
                                                 (short)0, acc, false, false);
  return acc;
}

// ---------------- prep: weight transpose to f16 ----------------
__global__ void prep_transpose(const float* __restrict__ Wap,
                               const float* __restrict__ Wce,
                               h16* __restrict__ WapT, h16* __restrict__ WceT) {
  const int n = blockIdx.x & 255;
  const int k = threadIdx.x;
  if (blockIdx.x < 256)
    WapT[n * 256 + k] = (h16)Wap[k * 256 + n];
  else
    WceT[n * 256 + k] = (h16)Wce[k * 256 + n];
}

// ---------------- prep: basal[t][j] = (gauss(t) @ Wbasal)[j] ----------------
__global__ void prep_basal(const float* __restrict__ Wbasal, float* __restrict__ basal) {
  const int t = blockIdx.x;       // 0..99  (x = t+1)
  const int j = threadIdx.x;      // 0..255
  const float x = (float)(t + 1);
  float acc = 0.0f;
  for (int i = 0; i < 256; ++i) {
    float c = (float)i * (100.0f / 255.0f);
    float d = (x - c) * (1.0f / 5.0f);
    acc += expf(-0.5f * d * d) * Wbasal[i * 256 + j];
  }
  basal[t * 256 + j] = acc;
}

// ---------------- main recurrent kernel ----------------
__global__ void __launch_bounds__(256, 1) rnn_kernel(
    const float* __restrict__ cue, const float* __restrict__ ec5_init,
    const h16* __restrict__ WapT, const h16* __restrict__ WceT,
    const float* __restrict__ basal, const float* __restrict__ bias,
    const float* __restrict__ Waction, float* __restrict__ out) {
  __shared__ h16 S[ROWS * 256];   // 64 KB, time-shared between ec3 and ca1

  const int tid  = threadIdx.x;
  const int lane = tid & 31;
  const int wave = tid >> 5;
  const int m    = lane & 15;     // N column / A row within tile
  const int hi   = lane >> 4;     // lane half select
  const int wrow = wave << 4;     // this wave's first local row
  const int gbase = blockIdx.x * ROWS + wrow;
  const int lbase = m * 256 + hi * 16;   // per-lane B-fragment base (elements)

  float biasv[NT];
#pragma unroll
  for (int nt = 0; nt < NT; ++nt) biasv[nt] = bias[nt * 16 + m];

  // State held in registers in C-fragment layout:
  // element (r, lane) of tile nt -> row gbase+hi*8+r, col nt*16+m
  v8f ec5v[NT], ec3v[NT];
  v16h a[KT];

  // ================= step 0 (peeled) =================
  // ec3(0)==0 -> GEMM1 degenerates: ca1_0 = relu(1.5*basal[0][col] - bias[col])
#pragma unroll
  for (int nt = 0; nt < NT; ++nt) {
    const int col = nt * 16 + m;
    const h16 ch = (h16)fmaxf(1.5f * basal[col] - biasv[nt], 0.0f);
#pragma unroll
    for (int r = 0; r < 8; ++r)
      S[sidx(wrow + hi * 8 + r, col)] = ch;
  }
  // GEMM2 for step 0: ec5 = clip(ec5_init + ca1_0 @ Wce); ec3 = clip(cue)
  load_a_frags(S, wrow, m, hi, a);
  {
    gh16* wce0 = opaque(WceT);
#pragma unroll
    for (int nt = 0; nt < NT; ++nt) {
      const int col = nt * 16 + m;
      v8f acc = gemm_tile(a, wce0, lbase + nt * 4096);
#pragma unroll
      for (int r = 0; r < 8; ++r) {
        const int gi = (gbase + hi * 8 + r) * 256 + col;
        float e5 = clip01(ec5_init[gi] + acc[r]);
        float e3 = clip01(cue[gi]);                // ec3_old == 0
        ec5v[nt][r] = e5;
        ec3v[nt][r] = e3;
        S[sidx(wrow + hi * 8 + r, col)] = (h16)e3;
      }
    }
  }

  // ================= steps 1..99 =================
  for (int t = 1; t < STEPS; ++t) {
    gh16* wap = opaque(WapT);   // fresh pointer each iter: no LICM/spill
    gh16* wce = opaque(WceT);

    // prefetch this step's basal row (per-lane columns)
    float basv[NT];
#pragma unroll
    for (int nt = 0; nt < NT; ++nt) basv[nt] = basal[t * 256 + nt * 16 + m];

    // ---- GEMM1: pre = ec3 @ Wapical; ca1 = relu(bas*(1+sigmoid(pre)) - bias)
    load_a_frags(S, wrow, m, hi, a);
#pragma unroll
    for (int nt = 0; nt < NT; ++nt) {
      const int col = nt * 16 + m;
      v8f acc = gemm_tile(a, wap, lbase + nt * 4096);
#pragma unroll
      for (int r = 0; r < 8; ++r) {
        float sg = 1.0f / (1.0f + __expf(-acc[r]));
        float c1 = fmaxf(basv[nt] * (1.0f + sg) - biasv[nt], 0.0f);
        S[sidx(wrow + hi * 8 + r, col)] = (h16)c1;   // ca1 overwrites ec3 image
      }
    }

    // ---- GEMM2: ec5 = clip(ec5 + ca1 @ Wca1ec5); ec3 = clip(ec5*ec3)
    load_a_frags(S, wrow, m, hi, a);                  // A = ca1 (own rows)
#pragma unroll
    for (int nt = 0; nt < NT; ++nt) {
      const int col = nt * 16 + m;
      v8f acc = gemm_tile(a, wce, lbase + nt * 4096);
#pragma unroll
      for (int r = 0; r < 8; ++r) {
        float e5 = clip01(ec5v[nt][r] + acc[r]);
        float e3 = clip01(e5 * ec3v[nt][r]);
        ec5v[nt][r] = e5;
        ec3v[nt][r] = e3;
        if (t != STEPS - 1)                           // keep ca1 on final step
          S[sidx(wrow + hi * 8 + r, col)] = (h16)e3;
      }
    }
  }

  __syncthreads();   // epilogue reads ca1 rows across waves

  // out = ca1_final @ Waction  (ACT=2): one (row, action) pair per thread
  {
    const int row = tid >> 1;
    const int act = tid & 1;
    float acc = 0.0f;
    for (int c = 0; c < 256; ++c)
      acc += (float)S[sidx(row, c)] * Waction[c * 2 + act];
    out[(blockIdx.x * ROWS + row) * 2 + act] = acc;
  }
}

extern "C" void kernel_launch(void* const* d_in, const int* in_sizes, int n_in,
                              void* d_out, int out_size, void* d_ws, size_t ws_size,
                              hipStream_t stream) {
  (void)in_sizes; (void)n_in; (void)out_size; (void)ws_size;
  const float* cue   = (const float*)d_in[0];
  const float* ec5i  = (const float*)d_in[1];
  const float* Wap   = (const float*)d_in[2];
  const float* Wbas  = (const float*)d_in[3];
  const float* Wce   = (const float*)d_in[4];
  const float* Wact  = (const float*)d_in[5];
  const float* bias  = (const float*)d_in[6];

  // workspace layout: WapT f16 [64K], WceT f16 [64K], basal f32 [100*256]
  h16*   WapT  = (h16*)d_ws;
  h16*   WceT  = WapT + 256 * 256;
  float* basal = (float*)(WceT + 256 * 256);
  float* out   = (float*)d_out;

  prep_transpose<<<512, 256, 0, stream>>>(Wap, Wce, WapT, WceT);
  prep_basal<<<100, 256, 0, stream>>>(Wbas, basal);
  rnn_kernel<<<BS / ROWS, 256, 0, stream>>>(cue, ec5i, WapT, WceT, basal, bias,
                                            Wact, out);
}
